// Speller_25658134627071
// MI455X (gfx1250) — matrix operand order repaired and checked by reading, not compiled, not run on hardware
//
#include <hip/hip_runtime.h>
#include <hip/hip_bf16.h>
#include <stdint.h>

// Problem constants (from reference): B=64, T=200, encT=1000, U=512, V=46
#define B_    64
#define T_    200
#define ENC_  1000
#define U_    512
#define V_    46
#define XPAD_ 1088          // x (2U+V = 1070) padded to multiple of 32
#define K0_   1600          // XPAD_ + U_ (concatenated [x | h0] GEMM)
#define K1_   1024          // U_ + U_   (concatenated [ctx | h1] GEMM)
#define N4_   2048          // 4U
#define VD_   48            // V padded to 48 for 16x16 tiles
#define ENC_TILES 63        // ceil(1000/16)

typedef __attribute__((ext_vector_type(16))) __bf16 v16bf;
typedef __attribute__((ext_vector_type(8)))  float  v8f;

union BfPack { uint4 q[2]; v16bf v; };

__device__ __forceinline__ unsigned short f2bf(float f) {
  uint32_t u = __float_as_uint(f);
  return (unsigned short)((u + 0x7FFFu + ((u >> 16) & 1u)) >> 16);  // RNE
}
__device__ __forceinline__ float bf2f(unsigned short s) {
  return __uint_as_float(((uint32_t)s) << 16);
}
__device__ __forceinline__ float sigm(float x) { return 1.0f / (1.0f + __expf(-x)); }

// ---------------------------------------------------------------------------
// Generic bf16 WMMA GEMM: C(MxN,f32) = A(MxK,bf16 row-major) * Bt(NxK,bf16)^T + bias
// One wave per 16x16 tile; K consumed in blocks of 32 via v_wmma_f32_16x16x32_bf16.
// A frag per lane (doc 7.12.2, 16-bit A 16x32):  row=lane&15, K = 16*(e>>3)+hf*8+(e&7)
//   -> two contiguous 8-element (16B) chunks at K offsets hf*8 and 16+hf*8.
// B frag per lane (16-bit B 32x16): col=lane&15, K = 16*(e>>3)+hf*8+(e&7) -> same as A:
//   two 16B chunks of the transposed weight row at K offsets hf*8 and 16+hf*8.
// ---------------------------------------------------------------------------
__global__ void wmma_gemm_bf16(const unsigned short* __restrict__ A, int lda,
                               const unsigned short* __restrict__ Bt, int ldb,
                               const float* __restrict__ bias,
                               float* __restrict__ C, int ldc,
                               int Mtiles, int Ntiles, int kblocks) {
  int wave = (int)((blockIdx.x * blockDim.x + threadIdx.x) >> 5);
  if (wave >= Mtiles * Ntiles) return;              // uniform per wave; EXEC stays full
  int mt = wave % Mtiles, nt = wave / Mtiles;
  int lane = (int)(threadIdx.x & 31);
  int hf = lane >> 4, r = lane & 15;
  const unsigned short* arow = A  + (size_t)(mt * 16 + r) * lda + hf * 8;
  const unsigned short* brow = Bt + (size_t)(nt * 16 + r) * ldb + hf * 8;
  v8f acc = {};
  for (int kb = 0; kb < kblocks; ++kb) {
    BfPack ua, ub;
    ua.q[0] = *(const uint4*)(arow);
    ua.q[1] = *(const uint4*)(arow + 16);
    ub.q[0] = *(const uint4*)(brow);
    ub.q[1] = *(const uint4*)(brow + 16);
    acc = __builtin_amdgcn_wmma_f32_16x16x32_bf16(false, ua.v, false, ub.v,
                                                  (short)0, acc, false, false);
    arow += 32; brow += 32;
  }
  int col = nt * 16 + r;
  float bv = bias ? bias[col] : 0.0f;
  int row0 = mt * 16 + hf * 8;                      // C/D layout: VGPR v -> M=row0+v
#pragma unroll
  for (int v = 0; v < 8; ++v) C[(size_t)(row0 + v) * ldc + col] = acc[v] + bv;
}

// ---------------------------------------------------------------------------
// Attention scores via WMMA: scores[b][t] = sum_u s0[b][u] * h[b][t][u]
// A tile = 16 encoder rows of h (bf16), B = s0 broadcast into all 16 columns
// (every output column equals the score; column 0 is extracted at store).
// ---------------------------------------------------------------------------
__global__ void attn_scores_wmma(const unsigned short* __restrict__ hbf,
                                 const unsigned short* __restrict__ s0,
                                 float* __restrict__ scores) {
  int wave = (int)((blockIdx.x * blockDim.x + threadIdx.x) >> 5);
  if (wave >= B_ * ENC_TILES) return;
  int b = wave / ENC_TILES, mt = wave % ENC_TILES;
  int lane = (int)(threadIdx.x & 31);
  int hf = lane >> 4, r = lane & 15;
  int row = mt * 16 + r; if (row > ENC_ - 1) row = ENC_ - 1;   // clamp tail tile
  const unsigned short* arow = hbf + ((size_t)b * ENC_ + row) * U_ + hf * 8;
  const unsigned short* brow = s0 + (size_t)b * U_ + hf * 8;
  v8f acc = {};
  for (int kb = 0; kb < U_ / 32; ++kb) {
    BfPack ua, ub;
    ua.q[0] = *(const uint4*)(arow);
    ua.q[1] = *(const uint4*)(arow + 16);
    ub.q[0] = *(const uint4*)(brow);            // identical across lanes: broadcast B
    ub.q[1] = *(const uint4*)(brow + 16);
    acc = __builtin_amdgcn_wmma_f32_16x16x32_bf16(false, ua.v, false, ub.v,
                                                  (short)0, acc, false, false);
    arow += 32; brow += 32;
  }
  if ((lane & 15) == 0) {                        // lanes 0 (rows 0-7) and 16 (rows 8-15)
    int t0 = mt * 16 + hf * 8;
#pragma unroll
    for (int v = 0; v < 8; ++v) {
      int tt = t0 + v;
      if (tt < ENC_) scores[(size_t)b * ENC_ + tt] = acc[v];
    }
  }
}

// softmax over encT=1000, in place; one block per batch row
__global__ void softmax_enc(float* __restrict__ s) {
  __shared__ float red[256];
  int b = blockIdx.x, tid = threadIdx.x;
  float* sb = s + (size_t)b * ENC_;
  float m = -1e30f;
  for (int i = tid; i < ENC_; i += 256) m = fmaxf(m, sb[i]);
  red[tid] = m; __syncthreads();
  for (int st = 128; st > 0; st >>= 1) { if (tid < st) red[tid] = fmaxf(red[tid], red[tid + st]); __syncthreads(); }
  m = red[0]; __syncthreads();
  float sum = 0.0f;
  for (int i = tid; i < ENC_; i += 256) { float e = __expf(sb[i] - m); sb[i] = e; sum += e; }
  red[tid] = sum; __syncthreads();
  for (int st = 128; st > 0; st >>= 1) { if (tid < st) red[tid] += red[tid + st]; __syncthreads(); }
  float inv = 1.0f / red[0];
  for (int i = tid; i < ENC_; i += 256) sb[i] *= inv;
}

// context[b][u] = sum_t attn[b][t] * h[b][t][u]; writes bf16 into three A-matrices
__global__ void attn_context(const float* __restrict__ attn,
                             const unsigned short* __restrict__ hbf,
                             unsigned short* __restrict__ A0ctx,   // A0 + 512, ld K0_
                             unsigned short* __restrict__ A1ctx,   // A1 + 0,   ld K1_
                             unsigned short* __restrict__ Adctx) { // Ad + 512, ld K1_
  __shared__ float aw[256];
  int b = blockIdx.x, u = threadIdx.x;            // 512 threads
  const unsigned short* hb = hbf + (size_t)b * ENC_ * U_ + u;
  float acc = 0.0f;
  for (int base = 0; base < ENC_; base += 256) {
    int n = ENC_ - base; if (n > 256) n = 256;
    if ((int)threadIdx.x < n) aw[threadIdx.x] = attn[(size_t)b * ENC_ + base + threadIdx.x];
    __syncthreads();
    for (int j = 0; j < n; ++j) acc = fmaf(aw[j], bf2f(hb[(size_t)(base + j) * U_]), acc);
    __syncthreads();
  }
  unsigned short cb = f2bf(acc);
  A0ctx[(size_t)b * K0_ + u] = cb;
  A1ctx[(size_t)b * K1_ + u] = cb;
  Adctx[(size_t)b * K1_ + u] = cb;
}

// embedding lookup -> bf16 into A0[:, 0:512]
__global__ void embed_x(const float* __restrict__ emb, const int* __restrict__ y,
                        int t, unsigned short* __restrict__ A0) {
  int b = blockIdx.x, u = threadIdx.x;            // 512 threads
  int tok = y[(size_t)b * T_ + t];
  A0[(size_t)b * K0_ + u] = f2bf(emb[(size_t)tok * U_ + u]);
}

// Keras LSTM pointwise: gates (i,f,g,o); writes new h (bf16) to two destinations
__global__ void lstm_point(const float* __restrict__ z, float* __restrict__ c,
                           unsigned short* __restrict__ d1, int ld1,
                           unsigned short* __restrict__ d2, int ld2) {
  int b = blockIdx.x, u = threadIdx.x;            // 512 threads
  const float* zb = z + (size_t)b * N4_;
  float iv = sigm(zb[u]);
  float fv = sigm(zb[U_ + u]);
  float gv = tanhf(zb[2 * U_ + u]);
  float ov = sigm(zb[3 * U_ + u]);
  float cn = fv * c[(size_t)b * U_ + u] + iv * gv;
  c[(size_t)b * U_ + u] = cn;
  unsigned short hb = f2bf(ov * tanhf(cn));
  d1[(size_t)b * ld1 + u] = hb;
  d2[(size_t)b * ld2 + u] = hb;
}

// log-softmax over V=46; writes output slice and y_hat (bf16) back into A0[:,1024:1070]
__global__ void lsm_out(const float* __restrict__ dist, float* __restrict__ out,
                        unsigned short* __restrict__ A0y, int t) {
  __shared__ float red[64];
  int b = blockIdx.x, v = threadIdx.x;            // 64 threads
  float val = (v < V_) ? dist[(size_t)b * VD_ + v] : -1e30f;
  red[v] = val; __syncthreads();
  for (int st = 32; st > 0; st >>= 1) { if (v < st) red[v] = fmaxf(red[v], red[v + st]); __syncthreads(); }
  float m = red[0]; __syncthreads();
  red[v] = (v < V_) ? __expf(val - m) : 0.0f; __syncthreads();
  for (int st = 32; st > 0; st >>= 1) { if (v < st) red[v] += red[v + st]; __syncthreads(); }
  float lse = m + logf(red[0]);
  if (v < V_) {
    float yv = val - lse;
    out[((size_t)b * T_ + t) * V_ + v] = yv;
    A0y[(size_t)b * K0_ + v] = f2bf(yv);
  }
}

// ---------------- one-time preprocessing (per call, deterministic) ----------------
__global__ void prep_wc0(const float* __restrict__ W0, const float* __restrict__ U0,
                         unsigned short* __restrict__ Wt) {   // (2048 x 1600) N-major
  int idx = blockIdx.x * blockDim.x + threadIdx.x;
  if (idx >= N4_ * K0_) return;
  int n = idx / K0_, k = idx % K0_;
  float v;
  if (k < 2 * U_ + V_)      v = W0[(size_t)k * N4_ + n];
  else if (k < XPAD_)       v = 0.0f;
  else                      v = U0[(size_t)(k - XPAD_) * N4_ + n];
  Wt[idx] = f2bf(v);
}
__global__ void prep_wc1(const float* __restrict__ W1, const float* __restrict__ U1,
                         unsigned short* __restrict__ Wt) {   // (2048 x 1024)
  int idx = blockIdx.x * blockDim.x + threadIdx.x;
  if (idx >= N4_ * K1_) return;
  int n = idx / K1_, k = idx % K1_;
  float v = (k < U_) ? W1[(size_t)k * N4_ + n] : U1[(size_t)(k - U_) * N4_ + n];
  Wt[idx] = f2bf(v);
}
__global__ void prep_wd(const float* __restrict__ Wd, const float* __restrict__ bd,
                        unsigned short* __restrict__ Wt, float* __restrict__ bdp) {
  int idx = blockIdx.x * blockDim.x + threadIdx.x;            // (48 x 1024)
  if (idx < VD_ * K1_) {
    int n = idx / K1_, k = idx % K1_;
    Wt[idx] = f2bf((n < V_) ? Wd[(size_t)k * V_ + n] : 0.0f);
  }
  if (idx < VD_) bdp[idx] = (idx < V_) ? bd[idx] : 0.0f;
}
__global__ void prep_hbf(const float* __restrict__ h, unsigned short* __restrict__ hbf, int n) {
  int i = blockIdx.x * blockDim.x + threadIdx.x;
  if (i < n) hbf[i] = f2bf(h[i]);
}
__global__ void zero_us(unsigned short* p, int n) {
  int i = blockIdx.x * blockDim.x + threadIdx.x;
  if (i < n) p[i] = 0;
}
__global__ void zero_f(float* p, int n) {
  int i = blockIdx.x * blockDim.x + threadIdx.x;
  if (i < n) p[i] = 0.0f;
}

// ---------------------------------------------------------------------------
extern "C" void kernel_launch(void* const* d_in, const int* in_sizes, int n_in,
                              void* d_out, int out_size, void* d_ws, size_t ws_size,
                              hipStream_t stream) {
  (void)in_sizes; (void)n_in; (void)out_size; (void)ws_size;
  const float* h   = (const float*)d_in[0];
  const int*   y   = (const int*)  d_in[1];
  const float* emb = (const float*)d_in[2];
  const float* W0  = (const float*)d_in[3];
  const float* U0  = (const float*)d_in[4];
  const float* b0  = (const float*)d_in[5];
  const float* W1  = (const float*)d_in[6];
  const float* U1  = (const float*)d_in[7];
  const float* b1  = (const float*)d_in[8];
  const float* Wd  = (const float*)d_in[9];
  const float* bd  = (const float*)d_in[10];
  float* out = (float*)d_out;

  char* ws = (char*)d_ws;
  size_t off = 0;
  auto carve = [&](size_t bytes) -> char* {
    off = (off + 255) & ~(size_t)255;
    char* p = ws + off; off += bytes; return p;
  };
  unsigned short* hbf  = (unsigned short*)carve((size_t)B_ * ENC_ * U_ * 2); // 65.5 MB (L2-resident)
  unsigned short* Wc0t = (unsigned short*)carve((size_t)N4_ * K0_ * 2);      // 6.55 MB
  unsigned short* Wc1t = (unsigned short*)carve((size_t)N4_ * K1_ * 2);      // 4.19 MB
  unsigned short* Wdt  = (unsigned short*)carve((size_t)VD_ * K1_ * 2);
  float*          bdp  = (float*)         carve((size_t)VD_ * 4);
  unsigned short* A0   = (unsigned short*)carve((size_t)B_ * K0_ * 2);       // [e|ctx|yhat|pad|h0]
  unsigned short* A1   = (unsigned short*)carve((size_t)B_ * K1_ * 2);       // [ctx|h1]
  unsigned short* Ad   = (unsigned short*)carve((size_t)B_ * K1_ * 2);       // [s1|ctx]
  unsigned short* s0bf = (unsigned short*)carve((size_t)B_ * U_ * 2);
  float*          z0   = (float*)carve((size_t)B_ * N4_ * 4);
  float*          z1   = (float*)carve((size_t)B_ * N4_ * 4);
  float*          c0   = (float*)carve((size_t)B_ * U_ * 4);
  float*          c1   = (float*)carve((size_t)B_ * U_ * 4);
  float*          attn = (float*)carve((size_t)B_ * ENC_ * 4);               // scores -> softmax in place
  float*          dist = (float*)carve((size_t)B_ * VD_ * 4);

  // ---- one-time prep: bf16 conversion / weight transposition / state zeroing ----
  {
    int n = B_ * ENC_ * U_;
    prep_hbf<<<(n + 255) / 256, 256, 0, stream>>>(h, hbf, n);
  }
  prep_wc0<<<(N4_ * K0_ + 255) / 256, 256, 0, stream>>>(W0, U0, Wc0t);
  prep_wc1<<<(N4_ * K1_ + 255) / 256, 256, 0, stream>>>(W1, U1, Wc1t);
  prep_wd <<<(VD_ * K1_ + 255) / 256, 256, 0, stream>>>(Wd, bd, Wdt, bdp);
  zero_us<<<(B_ * K0_ + 255) / 256, 256, 0, stream>>>(A0, B_ * K0_); // ctx=yhat=h0=pad=0
  zero_us<<<(B_ * K1_ + 255) / 256, 256, 0, stream>>>(A1, B_ * K1_);
  zero_us<<<(B_ * K1_ + 255) / 256, 256, 0, stream>>>(Ad, B_ * K1_);
  zero_f <<<(B_ * U_ + 255) / 256, 256, 0, stream>>>(c0, B_ * U_);
  zero_f <<<(B_ * U_ + 255) / 256, 256, 0, stream>>>(c1, B_ * U_);

  const int g0_tiles = (B_ / 16) * (N4_ / 16);     // 512 tiles
  const int sc_tiles = B_ * ENC_TILES;             // 4032 tiles
  const int gd_tiles = (B_ / 16) * (VD_ / 16);     // 12 tiles

  // ---- sequential decode over T steps ----
  for (int t = 0; t < T_; ++t) {
    // x_t = [emb[y_t] | ctx | y_hat] (ctx/y_hat/h0 already resident in A0)
    embed_x<<<B_, U_, 0, stream>>>(emb, y, t, A0);
    // z0 = [x|h0] @ [W0;U0] + b0
    wmma_gemm_bf16<<<g0_tiles / 8, 256, 0, stream>>>(A0, K0_, Wc0t, K0_, b0, z0, N4_,
                                                     B_ / 16, N4_ / 16, K0_ / 32);
    // LSTM0 pointwise -> s0 (for attention) and h0 tail of A0 (for next step)
    lstm_point<<<B_, U_, 0, stream>>>(z0, c0, s0bf, U_, A0 + XPAD_, K0_);
    // attention
    attn_scores_wmma<<<sc_tiles / 8, 256, 0, stream>>>(hbf, s0bf, attn);
    softmax_enc<<<B_, 256, 0, stream>>>(attn);
    attn_context<<<B_, U_, 0, stream>>>(attn, hbf, A0 + U_, A1, Ad + U_);
    // z1 = [ctx|h1] @ [W1;U1] + b1
    wmma_gemm_bf16<<<g0_tiles / 8, 256, 0, stream>>>(A1, K1_, Wc1t, K1_, b1, z1, N4_,
                                                     B_ / 16, N4_ / 16, K1_ / 32);
    // LSTM1 pointwise -> s1 head of Ad (decoder input) and h1 tail of A1 (next step)
    lstm_point<<<B_, U_, 0, stream>>>(z1, c1, Ad, K1_, A1 + U_, K1_);
    // dist = [s1|ctx] @ Wd + bd
    wmma_gemm_bf16<<<(gd_tiles + 7) / 8, 256, 0, stream>>>(Ad, K1_, Wdt, K1_, bdp, dist, VD_,
                                                           B_ / 16, VD_ / 16, K1_ / 32);
    // log-softmax -> output slice + y_hat back into A0
    lsm_out<<<B_, 64, 0, stream>>>(dist, out, A0 + 2 * U_, t);
  }
}
